// CNN_tagger_20761871909366
// MI455X (gfx1250) — matrix operand order repaired and checked by reading, not compiled, not run on hardware
//
#include <hip/hip_runtime.h>
#include <hip/hip_bf16.h>

// ---------------------------------------------------------------------------
// CNN tagger for MI455X (gfx1250), wave32 + v_wmma_f32_16x16x32_bf16.
//
// Implicit-GEMM conv: branch k has K-dim = k*512 (tap-major), M = 256 output
// channels, N = 16 seq positions per workgroup. feats (768 x 16) live only in
// LDS; linear (64x768) fused in the same kernel.
// All global accesses go through explicit addrspace(1) pointers so they lower
// to global_load/store (LOADcnt/STOREcnt only) instead of flat (which would
// also tie up DScnt and serialize against the LDS B-fragment stream).
// ---------------------------------------------------------------------------

typedef __attribute__((ext_vector_type(16))) __bf16 v16bf;
typedef __attribute__((ext_vector_type(8)))  float  v8f;
typedef __attribute__((ext_vector_type(4)))  int    v4i;
typedef __attribute__((ext_vector_type(4)))  float  v4f;
typedef __attribute__((ext_vector_type(2)))  int    v2i;

typedef const v4i __attribute__((address_space(1)))* gcv4i;
typedef const v4f __attribute__((address_space(1)))* gcv4f;
typedef       v4f __attribute__((address_space(1)))* gv4f;

static __device__ __forceinline__ v4i gload4i(const void* p) {
  return *(gcv4i)(unsigned long long)p;
}
static __device__ __forceinline__ v4f gload4f(const void* p) {
  return *(gcv4f)(unsigned long long)p;
}
static __device__ __forceinline__ void gstore4f(void* p, v4f v) {
  *(gv4f)(unsigned long long)p = v;
}

union Frag16 { v4i q[2]; v16bf bf; };      // 16 bf16 = 32B A/B operand
union Pack8  { __bf16 p[8]; v4i q; };      // 8 bf16  = 16B ds_store_b128
union Pack4  { __bf16 p[4]; v2i q; };      // 4 bf16  = 8B  ds_store_b64

#define XSTR 520    // padded bf16 row stride for x tile  (520*2/4 % 64 == 4)
#define FSTR 776    // padded bf16 row stride for featsT  (776*2/4 % 64 == 4)

// ---- weight repack: fp32 [NK][D][k] -> bf16 [NK][k*D] (K = dk*512 + d) ----
__global__ void pack_conv_w(const float* __restrict__ src,
                            __bf16* __restrict__ dst, int k) {
  int n = 256 * 512 * k;
  for (int t = blockIdx.x * blockDim.x + threadIdx.x; t < n;
       t += gridDim.x * blockDim.x) {
    int nk  = t / (512 * k);
    int rem = t - nk * 512 * k;
    int dk  = rem >> 9;          // /512
    int d   = rem & 511;
    dst[t] = (__bf16)src[(nk * 512 + d) * k + dk];
  }
}

__global__ void pack_lin_w(const float* __restrict__ src,
                           __bf16* __restrict__ dst, int n) {
  int t = blockIdx.x * blockDim.x + threadIdx.x;
  if (t < n) dst[t] = (__bf16)src[t];
}

// ---- fused conv(k=3,5,7) + relu + linear(768->64) -------------------------
__global__ __launch_bounds__(256) void cnn_tagger_kernel(
    const float* __restrict__ x,            // [32][512][512]
    const float* __restrict__ cb3, const float* __restrict__ cb5,
    const float* __restrict__ cb7, const float* __restrict__ linb,
    const __bf16* __restrict__ wA3,         // [256][3*512] bf16
    const __bf16* __restrict__ wA5,         // [256][5*512]
    const __bf16* __restrict__ wA7,         // [256][7*512]
    const __bf16* __restrict__ wL,          // [64][768]
    float* __restrict__ out)                // [32][512][64]
{
  __shared__ __bf16 lds_x[22 * XSTR];       // x tile (16 s + 3 halo each side)
  __shared__ __bf16 feats[16 * FSTR];       // featsT[n][c], c in [0,768)

  const int tid  = threadIdx.x;
  const int wave = tid >> 5;
  const int lane = tid & 31;
  const int i    = lane & 15;               // row/col selector within tile
  const int h    = lane >> 4;               // half-wave selector

  const int b  = blockIdx.x >> 5;           // batch
  const int s0 = (blockIdx.x & 31) << 4;    // seq tile start

  // -------- Phase 1: x[b, s0-3 .. s0+18, :] -> LDS (bf16, zero-padded) ----
  const float* xb = x + (size_t)b * (512 * 512);
  for (int it = 0; it < 11; ++it) {         // 22 rows * 128 float4 = 2816
    int f   = it * 256 + tid;
    int row = f >> 7;
    int c4  = f & 127;
    int sg  = s0 - 3 + row;
    v4f v = {0.f, 0.f, 0.f, 0.f};
    if (sg >= 0 && sg < 512) v = gload4f(xb + sg * 512 + c4 * 4);
    Pack4 pk;
    pk.p[0] = (__bf16)v.x; pk.p[1] = (__bf16)v.y;
    pk.p[2] = (__bf16)v.z; pk.p[3] = (__bf16)v.w;
    *(v2i*)&lds_x[row * XSTR + c4 * 4] = pk.q;
  }
  __syncthreads();

  // -------- Phase 2: three conv branches as implicit GEMM -----------------
  // Per branch, each wave owns channels [wave*16, +16) and [wave*16+128, +16)
  // so both M-tiles share one B fragment per K-chunk.
  const int     ks[3]  = {3, 5, 7};
  const __bf16* wAs[3] = {wA3, wA5, wA7};
  const float*  cbs[3] = {cb3, cb5, cb7};

  for (int bi = 0; bi < 3; ++bi) {
    const int k      = ks[bi];
    const int rowoff = 3 - ((k - 1) >> 1);  // halo alignment per branch
    const int Kb     = k * 512;
    const int nchunk = k * 16;              // K / 32
    const __bf16* wA = wAs[bi];
    const float*  cb = cbs[bi];

    const int cl0 = wave * 16;              // tile0 channel base (tile1: +128)
    const __bf16* w0 = wA + (size_t)(cl0 + i) * Kb + h * 8;
    const __bf16* w1 = w0 + (size_t)128 * Kb;

    v8f acc0 = {};
    v8f acc1 = {};
    for (int kc = 0; kc < nchunk; ++kc) {
      const int dk    = kc >> 4;            // tap index (512/32 chunks/tap)
      const int dbase = (kc & 15) << 5;
      Frag16 Bf;                            // B: x patch, 32 x 16 (shared)
      const __bf16* bp = &lds_x[(i + dk + rowoff) * XSTR + dbase + h * 8];
      Bf.q[0] = *(const v4i*)bp;
      Bf.q[1] = *(const v4i*)(bp + 16);
      Frag16 A0, A1;                        // A: weights, 16 x 32
      const __bf16* a0 = w0 + kc * 32;
      const __bf16* a1 = w1 + kc * 32;
      A0.q[0] = gload4i(a0); A0.q[1] = gload4i(a0 + 16);
      A1.q[0] = gload4i(a1); A1.q[1] = gload4i(a1 + 16);
      acc0 = __builtin_amdgcn_wmma_f32_16x16x32_bf16(
          false, A0.bf, false, Bf.bf, (short)0, acc0, false, false);
      acc1 = __builtin_amdgcn_wmma_f32_16x16x32_bf16(
          false, A1.bf, false, Bf.bf, (short)0, acc1, false, false);
    }

    // bias + relu, pack to bf16, store transposed into featsT[n][c]
#pragma unroll
    for (int tl = 0; tl < 2; ++tl) {
      const v8f acc = tl ? acc1 : acc0;
      const int cl  = cl0 + tl * 128;
      v4f bl = gload4f(cb + cl + h * 8);
      v4f bh = gload4f(cb + cl + h * 8 + 4);
      Pack8 pk;
#pragma unroll
      for (int r = 0; r < 8; ++r) {
        float fv = acc[r] + ((r < 4) ? bl[r] : bh[r - 4]);
        fv = fv > 0.f ? fv : 0.f;
        pk.p[r] = (__bf16)fv;
      }
      *(v4i*)&feats[i * FSTR + bi * 256 + cl + h * 8] = pk.q;
    }
  }
  __syncthreads();

  // -------- Phase 3: linear 64x768 @ featsT (waves 0..3) ------------------
  if (wave < 4) {
    const int o0 = wave * 16;
    const __bf16* wrow = wL + (size_t)(o0 + i) * 768 + h * 8;
    v8f acc = {};
    for (int kc = 0; kc < 24; ++kc) {
      Frag16 Bf;
      const __bf16* bp = &feats[i * FSTR + kc * 32 + h * 8];
      Bf.q[0] = *(const v4i*)bp;
      Bf.q[1] = *(const v4i*)(bp + 16);
      Frag16 Af;
      const __bf16* ap = wrow + kc * 32;
      Af.q[0] = gload4i(ap);
      Af.q[1] = gload4i(ap + 16);
      acc = __builtin_amdgcn_wmma_f32_16x16x32_bf16(
          false, Af.bf, false, Bf.bf, (short)0, acc, false, false);
    }
    v4f b0 = gload4f(linb + o0 + h * 8);
    v4f b1 = gload4f(linb + o0 + h * 8 + 4);
    v4f r0, r1;
    r0.x = acc[0] + b0.x; r0.y = acc[1] + b0.y;
    r0.z = acc[2] + b0.z; r0.w = acc[3] + b0.w;
    r1.x = acc[4] + b1.x; r1.y = acc[5] + b1.y;
    r1.z = acc[6] + b1.z; r1.w = acc[7] + b1.w;
    float* op = out + ((size_t)(b * 512 + s0 + i)) * 64 + o0 + h * 8;
    gstore4f(op, r0);
    gstore4f(op + 4, r1);
  }
}

// ---------------------------------------------------------------------------
extern "C" void kernel_launch(void* const* d_in, const int* in_sizes, int n_in,
                              void* d_out, int out_size, void* d_ws,
                              size_t ws_size, hipStream_t stream) {
  const float* x   = (const float*)d_in[0];
  const float* w3  = (const float*)d_in[1];
  const float* b3  = (const float*)d_in[2];
  const float* w5  = (const float*)d_in[3];
  const float* b5  = (const float*)d_in[4];
  const float* w7  = (const float*)d_in[5];
  const float* b7  = (const float*)d_in[6];
  const float* lw  = (const float*)d_in[7];
  const float* lb  = (const float*)d_in[8];
  float* out = (float*)d_out;

  // ws layout (bf16): A3 | A5 | A7 | lin   (~3.85 MB total)
  __bf16* wA3 = (__bf16*)d_ws;
  __bf16* wA5 = wA3 + 256 * (3 * 512);
  __bf16* wA7 = wA5 + 256 * (5 * 512);
  __bf16* wL  = wA7 + 256 * (7 * 512);

  pack_conv_w<<<(256 * 512 * 3 + 255) / 256, 256, 0, stream>>>(w3, wA3, 3);
  pack_conv_w<<<(256 * 512 * 5 + 255) / 256, 256, 0, stream>>>(w5, wA5, 5);
  pack_conv_w<<<(256 * 512 * 7 + 255) / 256, 256, 0, stream>>>(w7, wA7, 7);
  pack_lin_w<<<(64 * 768 + 255) / 256, 256, 0, stream>>>(lw, wL, 64 * 768);

  // one workgroup per (batch, 16-seq tile): 32 * 32 = 1024 WGs, 8 waves each
  cnn_tagger_kernel<<<1024, 256, 0, stream>>>(x, b3, b5, b7, lb,
                                              wA3, wA5, wA7, wL, out);
}
